// LorentzAttention_23381801959961
// MI455X (gfx1250) — matrix operand order repaired and checked by reading, not compiled, not run on hardware
//
#include <hip/hip_runtime.h>

// ---------------------------------------------------------------------------
// Lorentz attention for MI455X (gfx1250, wave32, WMMA 16x16x32 f16).
// B=4, N=2048, S=256, H=8, hs=32. Attention padded to d_att=64 with the
// Lorentz time term folded into dim 32 (q_t vs -k_t), SCALE folded into Q.
// Softmax row reductions: single-instruction fused DPP16 ops
// (v_max_num_f32_dpp / v_add_f32_dpp with row_ror) -- no LDS, no canonicalize.
// ---------------------------------------------------------------------------

typedef __attribute__((ext_vector_type(16))) _Float16 v16h;
typedef __attribute__((ext_vector_type(8)))  _Float16 v8h;
typedef __attribute__((ext_vector_type(8)))  float    v8f;

#define B_   4
#define N_   2048
#define S_   256
#define H_   8
#define HS_  32
#define DA_  64   // padded attention dim (32 space + 1 time + 31 zeros)

__device__ __constant__ const float kInvCurv = 10.0f;                 // 1/CURV
__device__ __constant__ const float kR       = 3.16227766016837933f;  // sqrt(10)
__device__ __constant__ const float kScale   = 0.17677669529663689f;  // 1/sqrt(32)
__device__ __constant__ const float kEps     = 1e-9f;

__device__ __forceinline__ v8f wmma_f32_16x16x32_f16(v16h a, v16h b, v8f c) {
  // 8 args: (neg_a, A, neg_b, B, c_mod, C, reuse_a, reuse_b)
  return __builtin_amdgcn_wmma_f32_16x16x32_f16(false, a, false, b, (short)0, c,
                                                false, false);
}

// ---- Fused DPP16 butterfly steps: dst = op(ror(src), src), one VALU op -----
#define DEF_DPP_RED(fn, inst, rot)                                          \
  __device__ __forceinline__ float fn(float v) {                            \
    float d;                                                                \
    asm(inst " %0, %1, %1 " rot                                             \
             " row_mask:0xf bank_mask:0xf bound_ctrl:1"                     \
        : "=v"(d) : "v"(v));                                                \
    return d;                                                               \
  }
DEF_DPP_RED(max_ror8, "v_max_num_f32_dpp", "row_ror:8")
DEF_DPP_RED(max_ror4, "v_max_num_f32_dpp", "row_ror:4")
DEF_DPP_RED(max_ror2, "v_max_num_f32_dpp", "row_ror:2")
DEF_DPP_RED(max_ror1, "v_max_num_f32_dpp", "row_ror:1")
DEF_DPP_RED(add_ror8, "v_add_f32_dpp", "row_ror:8")
DEF_DPP_RED(add_ror4, "v_add_f32_dpp", "row_ror:4")
DEF_DPP_RED(add_ror2, "v_add_f32_dpp", "row_ror:2")
DEF_DPP_RED(add_ror1, "v_add_f32_dpp", "row_ror:1")

__device__ __forceinline__ float row_max16(float v) {
  v = max_ror8(v); v = max_ror4(v); v = max_ror2(v); v = max_ror1(v);
  return v;
}
__device__ __forceinline__ float row_sum16(float v) {
  v = add_ror8(v); v = add_ror4(v); v = add_ror2(v); v = add_ror1(v);
  return v;
}
// Plain max without clang's sNaN canonicalize companions.
__device__ __forceinline__ float fmax_raw(float a, float b) {
  float d;
  asm("v_max_num_f32_e32 %0, %1, %2" : "=v"(d) : "v"(a), "v"(b));
  return d;
}

// Full-wave reduction (cold paths only).
__device__ __forceinline__ float wave_sum32(float v) {
  v += __shfl_xor(v, 16, 32);
  v += __shfl_xor(v, 8, 32);
  v += __shfl_xor(v, 4, 32);
  v += __shfl_xor(v, 2, 32);
  v += __shfl_xor(v, 1, 32);
  return v;
}

// ---------------------------------------------------------------------------
// Kernel 1: QKV projection + Lorentz time / logmap0, emit WMMA-ready f16.
//   Qh[bh][n][0:32]=SCALE*qs, [32]=SCALE*qt, [33:64)=0
//   Kh[bh][n][0:32]=ks,       [32]=-kt,      [33:64)=0
//   VtT[bh][d][n] = logmap0_space(v)  (transposed for contiguous B-fragments)
// One block per (b,n); 8 waves x 3 tasks = 24 (head,comp) matvecs of 257x32.
// ---------------------------------------------------------------------------
__global__ __launch_bounds__(256) void lorentz_qkv_kernel(
    const float* __restrict__ x,
    const float* __restrict__ Wq, const float* __restrict__ bq,
    const float* __restrict__ Wk, const float* __restrict__ bk,
    const float* __restrict__ Wv, const float* __restrict__ bv,
    _Float16* __restrict__ Qh, _Float16* __restrict__ Kh,
    _Float16* __restrict__ VtT)
{
  __shared__ float xr[S_ + 1];
  const int row  = blockIdx.x;            // b*N + n
  const int b    = row / N_;
  const int n    = row - b * N_;
  const int tid  = threadIdx.x;
  const int lane = tid & 31;
  const int warp = tid >> 5;

  for (int i = tid; i < S_ + 1; i += 256) xr[i] = x[(size_t)row * (S_ + 1) + i];
  __syncthreads();

  for (int t = warp; t < 24; t += 8) {
    const int h    = t & 7;
    const int comp = t >> 3;               // 0=q, 1=k, 2=v
    const float* W  = (comp == 0) ? Wq : ((comp == 1) ? Wk : Wv);
    const float* bb = (comp == 0) ? bq : ((comp == 1) ? bk : bv);
    const float* pw = W + (size_t)h * (S_ + 1) * HS_ + lane;

    float acc = bb[h * HS_ + lane];
#pragma unroll 4
    for (int d = 0; d < S_ + 1; ++d) acc += xr[d] * pw[(size_t)d * HS_];

    const float ss = wave_sum32(acc * acc);          // |space|^2, all lanes
    const size_t rb = (size_t)(b * H_ + h) * N_ + n;

    if (comp == 0) {
      const float qt = sqrtf(kInvCurv + ss);
      _Float16* q = Qh + rb * DA_;
      q[lane]        = (_Float16)(kScale * acc);
      q[HS_ + lane]  = (_Float16)((lane == 0) ? kScale * qt : 0.0f);
    } else if (comp == 1) {
      const float kt = sqrtf(kInvCurv + ss);
      _Float16* kk = Kh + rb * DA_;
      kk[lane]       = (_Float16)acc;
      kk[HS_ + lane] = (_Float16)((lane == 0) ? -kt : 0.0f);
    } else {
      const float vt = sqrtf(kInvCurv + ss);
      const float vn = sqrtf(fmaxf(ss, kEps));
      const float dg = kR * acoshf(fmaxf(vt / kR, 1.0f + 1e-7f));
      VtT[((size_t)(b * H_ + h) * HS_ + lane) * N_ + n] =
          (_Float16)(dg * acc / vn);
    }
  }
}

// ---------------------------------------------------------------------------
// Kernel 2: fused flash attention + expmap0 scaling.
// Block = 128 threads (4 waves), each wave owns 16 query rows; key tiles of 32.
// A-fragment (16-bit 16x32): lane l -> row M=l&15, K = (l>=16?16:0) + e.
// B-fragment (16-bit 32x16): lane l -> col N=l&15, K = (l>=16?16:0) + e.
// C/D fragment (f32 16x16):  lane l, vgpr r -> M = r + (l>=16?8:0), N = l&15.
// A C-fragment row lives exactly in one DPP "row" (16 lanes) -> DPP reductions.
// ---------------------------------------------------------------------------
__global__ __launch_bounds__(128) void lorentz_attn_kernel(
    const _Float16* __restrict__ Qh, const _Float16* __restrict__ Kh,
    const _Float16* __restrict__ VtT, float* __restrict__ spc)
{
  __shared__ _Float16 pbuf[4 * 16 * 32];   // per-wave P staging (C -> A layout)

  const int tid  = threadIdx.x;
  const int lane = tid & 31;
  const int warp = tid >> 5;
  const int hlf  = lane >> 4;              // half-wave index
  const int ln   = lane & 15;

  const int tilesPer = N_ / 64;
  const int bh    = blockIdx.x / tilesPer;
  const int qtile = (blockIdx.x - bh * tilesPer) * 64 + warp * 16;

  const _Float16* Q = Qh  + (size_t)bh * N_ * DA_;
  const _Float16* K = Kh  + (size_t)bh * N_ * DA_;
  const _Float16* V = VtT + (size_t)bh * HS_ * N_;

  union AFrag { v16h v; v8h p[2]; };
  AFrag aq0, aq1, ap;

  {  // Q A-fragments: K 0..31 and 32..63 (two 16B chunks each)
    const _Float16* qp = Q + (size_t)(qtile + ln) * DA_ + hlf * 16;
    aq0.p[0] = *(const v8h*)(qp);
    aq0.p[1] = *(const v8h*)(qp + 8);
    aq1.p[0] = *(const v8h*)(qp + 32);
    aq1.p[1] = *(const v8h*)(qp + 40);
  }

  float m[8], lsum[8];
#pragma unroll
  for (int r = 0; r < 8; ++r) { m[r] = -3.0e38f; lsum[r] = 0.0f; }
  v8f acc0 = {0.f, 0.f, 0.f, 0.f, 0.f, 0.f, 0.f, 0.f};
  v8f acc1 = {0.f, 0.f, 0.f, 0.f, 0.f, 0.f, 0.f, 0.f};

  _Float16* P = pbuf + warp * 512;

  for (int j = 0; j < N_; j += 32) {
    // K tile as B-fragments (scores), one 32B contiguous load per fragment
    const _Float16* kp0 = K + (size_t)(j + ln) * DA_ + hlf * 16;
    const _Float16* kp1 = K + (size_t)(j + 16 + ln) * DA_ + hlf * 16;
    const v16h bk00 = *(const v16h*)(kp0);        // keys j..j+15,  d 0..31
    const v16h bk01 = *(const v16h*)(kp0 + 32);   // keys j..j+15,  d 32..63
    const v16h bk10 = *(const v16h*)(kp1);        // keys j+16..31, d 0..31
    const v16h bk11 = *(const v16h*)(kp1 + 32);   // keys j+16..31, d 32..63
    // V tile as B-fragments (P·V), contiguous thanks to transposed VtT
    const v16h bv0 = *(const v16h*)(V + (size_t)ln * N_ + j + hlf * 16);
    const v16h bv1 = *(const v16h*)(V + (size_t)(16 + ln) * N_ + j + hlf * 16);
    if (j + 32 < N_)
      __builtin_prefetch(K + (size_t)(j + 32 + ln) * DA_, 0, 1);

    v8f s0 = {0.f, 0.f, 0.f, 0.f, 0.f, 0.f, 0.f, 0.f};
    v8f s1 = {0.f, 0.f, 0.f, 0.f, 0.f, 0.f, 0.f, 0.f};
    s0 = wmma_f32_16x16x32_f16(aq0.v, bk00, s0);
    s0 = wmma_f32_16x16x32_f16(aq1.v, bk01, s0);
    s1 = wmma_f32_16x16x32_f16(aq0.v, bk10, s1);
    s1 = wmma_f32_16x16x32_f16(aq1.v, bk11, s1);

    // Online softmax: row = r + 8*hlf lives in the 16 lanes of one DPP row.
    float p0[8], p1[8];
#pragma unroll
    for (int r = 0; r < 8; ++r) {
      const float mx = row_max16(fmax_raw(s0[r], s1[r]));
      const float mn = fmax_raw(m[r], mx);
      const float sc = __expf(m[r] - mn);
      const float e0 = __expf(s0[r] - mn);
      const float e1 = __expf(s1[r] - mn);
      const float rs = row_sum16(e0 + e1);
      lsum[r] = lsum[r] * sc + rs;
      acc0[r] *= sc;
      acc1[r] *= sc;
      m[r] = mn;
      p0[r] = e0;
      p1[r] = e1;
    }

    // C-layout -> A-layout for P via LDS (row-major 16x32 f16)
    __syncthreads();
#pragma unroll
    for (int r = 0; r < 8; ++r) {
      P[(r + 8 * hlf) * 32 + ln]      = (_Float16)p0[r];
      P[(r + 8 * hlf) * 32 + 16 + ln] = (_Float16)p1[r];
    }
    __syncthreads();
    {
      const _Float16* pr = P + ln * 32 + hlf * 16;
      ap.p[0] = *(const v8h*)(pr);
      ap.p[1] = *(const v8h*)(pr + 8);
    }
    acc0 = wmma_f32_16x16x32_f16(ap.v, bv0, acc0);
    acc1 = wmma_f32_16x16x32_f16(ap.v, bv1, acc1);
  }

  // Epilogue: normalize, expmap0 factor (per-head time is dead in reference),
  // scatter into space_cat layout (B, N, H*32).
  const int b = bh / H_;
  const int h = bh - b * H_;
#pragma unroll
  for (int r = 0; r < 8; ++r) {
    const int qrow = qtile + r + 8 * hlf;
    const float inv = 1.0f / lsum[r];
    const float u0 = acc0[r] * inv;
    const float u1 = acc1[r] * inv;
    const float ss2 = row_sum16(u0 * u0 + u1 * u1);    // |u|^2 over 32 dims
    const float un  = sqrtf(fmaxf(ss2, kEps * kEps));
    const float fac = kR * sinhf(un / kR) / un;
    const size_t base = ((size_t)(b * N_ + qrow)) * S_ + h * HS_;
    spc[base + ln]      = fac * u0;
    spc[base + 16 + ln] = fac * u1;
  }
}

// ---------------------------------------------------------------------------
// Kernel 3: final Lorentz FC: row257 = [time(space_cat), space_cat],
// y = row257 @ Wo + bo, out = [time(y), y].  One block per (b,n).
// ---------------------------------------------------------------------------
__global__ __launch_bounds__(256) void lorentz_out_kernel(
    const float* __restrict__ spc, const float* __restrict__ Wo,
    const float* __restrict__ bo, float* __restrict__ out)
{
  __shared__ float xr[S_ + 1];
  __shared__ float red[8];
  const int row  = blockIdx.x;            // b*N + n
  const int tid  = threadIdx.x;
  const int lane = tid & 31;
  const int warp = tid >> 5;

  const float v = spc[(size_t)row * S_ + tid];
  xr[1 + tid] = v;
  const float ss = wave_sum32(v * v);
  if (lane == 0) red[warp] = ss;
  __syncthreads();
  if (tid == 0) {
    float t = 0.f;
#pragma unroll
    for (int i = 0; i < 8; ++i) t += red[i];
    xr[0] = sqrtf(kInvCurv + t);
  }
  __syncthreads();

  float acc = bo[tid];
  const float* pw = Wo + tid;             // Wo: (257, 256) row-major
#pragma unroll 4
  for (int d = 0; d < S_ + 1; ++d) acc += xr[d] * pw[(size_t)d * S_];

  const float ys = wave_sum32(acc * acc);
  if (lane == 0) red[warp] = ys;
  __syncthreads();
  if (tid == 0) {
    float t = 0.f;
#pragma unroll
    for (int i = 0; i < 8; ++i) t += red[i];
    out[(size_t)row * (S_ + 1)] = sqrtf(kInvCurv + t);
  }
  out[(size_t)row * (S_ + 1) + 1 + tid] = acc;
}

// ---------------------------------------------------------------------------
extern "C" void kernel_launch(void* const* d_in, const int* in_sizes, int n_in,
                              void* d_out, int out_size, void* d_ws,
                              size_t ws_size, hipStream_t stream) {
  (void)in_sizes; (void)n_in; (void)out_size; (void)ws_size;
  const float* x  = (const float*)d_in[0];
  const float* Wq = (const float*)d_in[1];
  const float* bq = (const float*)d_in[2];
  const float* Wk = (const float*)d_in[3];
  const float* bk = (const float*)d_in[4];
  const float* Wv = (const float*)d_in[5];
  const float* bv = (const float*)d_in[6];
  const float* Wo = (const float*)d_in[7];
  const float* bo = (const float*)d_in[8];

  // Workspace layout: Qh (8MB f16) | Kh (8MB f16) | VtT (4MB f16) | spc (8MB f32)
  const size_t QH_ELTS = (size_t)B_ * H_ * N_ * DA_;
  const size_t VT_ELTS = (size_t)B_ * H_ * HS_ * N_;
  _Float16* Qh  = (_Float16*)d_ws;
  _Float16* Kh  = Qh + QH_ELTS;
  _Float16* VtT = Kh + QH_ELTS;
  float*    spc = (float*)(VtT + VT_ELTS);

  lorentz_qkv_kernel<<<dim3(B_ * N_), dim3(256), 0, stream>>>(
      x, Wq, bq, Wk, bk, Wv, bv, Qh, Kh, VtT);
  lorentz_attn_kernel<<<dim3(B_ * H_ * (N_ / 64)), dim3(128), 0, stream>>>(
      Qh, Kh, VtT, spc);
  lorentz_out_kernel<<<dim3(B_ * N_), dim3(256), 0, stream>>>(
      spc, Wo, bo, (float*)d_out);
}
